// HeteroGConvGRU_73873437491317
// MI455X (gfx1250) — matrix operand order repaired and checked by reading, not compiled, not run on hardware
//
#include <hip/hip_runtime.h>
#include <hip/hip_bf16.h>
#include <cstdint>

typedef __attribute__((ext_vector_type(16))) __bf16 v16bf;
typedef __attribute__((ext_vector_type(8)))  float  v8f;

#define NN        50000
#define DD        128
#define ND        (NN*DD)
#define EE        800000
#define MTILES    3128            // 391 blocks * 8 mtiles; rows padded to 50048
#define NROWS_PAD (MTILES*16)
#define KC_TOT    16              // K = 512 in chunks of 32

#if __has_builtin(__builtin_amdgcn_tensor_load_to_lds) && __has_builtin(__builtin_amdgcn_s_wait_tensorcnt)
#define HAVE_TDM 1
typedef __attribute__((ext_vector_type(4))) unsigned int tdm_v4u;
typedef __attribute__((ext_vector_type(8))) int          tdm_v8i;
typedef __attribute__((ext_vector_type(4))) int          tdm_v4i;
#else
#define HAVE_TDM 0
#endif

__device__ __forceinline__ unsigned short f2bf(float f) {
    unsigned u = __float_as_uint(f);
    u += 0x7FFFu + ((u >> 16) & 1u);          // round to nearest even
    return (unsigned short)(u >> 16);
}
__device__ __forceinline__ float bf2f(unsigned short h) {
    return __uint_as_float(((unsigned)h) << 16);
}
__device__ __forceinline__ void atomicAddF(float* a, float v) {
    __hip_atomic_fetch_add(a, v, __ATOMIC_RELAXED, __HIP_MEMORY_SCOPE_AGENT);
}

#if HAVE_TDM
// Build a D# (ISA 8.3/8.4) for a 2-D dword tile and issue TENSOR_LOAD_TO_LDS.
// tile_d0_dw: row length in dwords; tile_d1: rows; stride_dw: memory row stride (dwords).
__device__ __forceinline__ void tdm_load_tile(unsigned long long gaddr, unsigned lds_byte,
                                              unsigned tile_d0_dw, unsigned tile_d1,
                                              unsigned stride_dw) {
    tdm_v4u g0 = { 1u,                                     // count=1, user descriptor
                   lds_byte,                               // lds_addr (bytes)
                   (unsigned)(gaddr & 0xFFFFFFFFull),
                   (unsigned)((gaddr >> 32) & 0x01FFFFFFull) | (2u << 30) }; // type=2 (image)
    tdm_v8i g1 = {
        (int)0x00020000u,                                          // wg_mask=0, data_size=4B
        (int)((tile_d0_dw & 0xFFFFu) << 16),                       // tensor_dim0[15:0]
        (int)((tile_d0_dw >> 16) | ((tile_d1 & 0xFFFFu) << 16)),   // tensor_dim0[31:16]|tensor_dim1[15:0]
        (int)((tile_d1 >> 16) | ((tile_d0_dw & 0xFFFFu) << 16)),   // tensor_dim1[31:16]|tile_dim0
        (int)(tile_d1 & 0xFFFFu),                                  // tile_dim1 | tile_dim2=0
        (int)stride_dw,                                            // tensor_dim0_stride[31:0]
        0, 0 };
    tdm_v4i gz4 = { 0, 0, 0, 0 };
    tdm_v8i gz8 = { 0, 0, 0, 0, 0, 0, 0, 0 };
    __builtin_amdgcn_tensor_load_to_lds(g0, g1, gz4, gz4, gz8, 0);
}
#endif

// ---------------- degree count ----------------
__global__ void deg_kernel(const int* __restrict__ dst, float* __restrict__ deg, int e) {
    int t = blockIdx.x * blockDim.x + threadIdx.x;
    if (t < e) atomicAddF(&deg[dst[t]], 1.0f);
}

// ---------------- scatter add (single feature) ----------------
__global__ void scatter_add_kernel(const float* __restrict__ feat,
                                   const int* __restrict__ src,
                                   const int* __restrict__ dst,
                                   float* __restrict__ acc, int e) {
    long long t = (long long)blockIdx.x * blockDim.x + threadIdx.x;
    if (t >= (long long)e * 32) return;
    int ed = (int)(t >> 5);
    int c4 = ((int)t & 31) * 4;
    int s = src[ed], d = dst[ed];
    const float4 v = *(const float4*)(feat + (size_t)s * DD + c4);
    float* o = acc + (size_t)d * DD + c4;
    atomicAddF(o + 0, v.x);
    atomicAddF(o + 1, v.y);
    atomicAddF(o + 2, v.z);
    atomicAddF(o + 3, v.w);
}

// ---------------- fused scatter add for two features over one edge list ----------------
__global__ void scatter_add2_kernel(const float* __restrict__ fa, const float* __restrict__ fb,
                                    const int* __restrict__ src, const int* __restrict__ dst,
                                    float* __restrict__ aa, float* __restrict__ ab, int e) {
    long long t = (long long)blockIdx.x * blockDim.x + threadIdx.x;
    if (t >= (long long)e * 32) return;
    int ed = (int)(t >> 5);
    int c4 = ((int)t & 31) * 4;
    int s = src[ed], d = dst[ed];
    const float4 va = *(const float4*)(fa + (size_t)s * DD + c4);
    const float4 vb = *(const float4*)(fb + (size_t)s * DD + c4);
    float* oa = aa + (size_t)d * DD + c4;
    float* ob = ab + (size_t)d * DD + c4;
    atomicAddF(oa + 0, va.x); atomicAddF(oa + 1, va.y);
    atomicAddF(oa + 2, va.z); atomicAddF(oa + 3, va.w);
    atomicAddF(ob + 0, vb.x); atomicAddF(ob + 1, vb.y);
    atomicAddF(ob + 2, vb.z); atomicAddF(ob + 3, vb.w);
}

// ---------------- pack A (node features) into WMMA bf16 hi/lo fragments ----------------
// A frag (16x32 bf16): lane = mrow + 16*khalf ; VGPR v = group*4 + within ;
// K = group*16 + khalf*8 + within*2 + parity. Frag index = (mt*16 + kc)*2 + hl.
__global__ void pack_A_kernel(const float* __restrict__ src, const float* __restrict__ deg,
                              unsigned int* __restrict__ ap32, int kchunk0, int nrows) {
    int t = blockIdx.x * blockDim.x + threadIdx.x;
    if (t >= NROWS_PAD * 64) return;
    int m  = t >> 6;
    int k  = (t & 63) * 2;                    // 0..126, covers 4 k-chunks
    float a0 = 0.f, a1 = 0.f;
    if (m < nrows) {
        a0 = src[(size_t)m * DD + k];
        a1 = src[(size_t)m * DD + k + 1];
        if (deg) { float c = fmaxf(deg[m], 1.0f); a0 /= c; a1 /= c; }
    }
    unsigned short h0 = f2bf(a0), h1 = f2bf(a1);
    unsigned short l0 = f2bf(a0 - bf2f(h0)), l1 = f2bf(a1 - bf2f(h1));
    int mt = m >> 4, mrow = m & 15;
    int kc = kchunk0 + (k >> 5);
    int kk = k & 31;
    int group = kk >> 4, r = kk & 15;
    int lane = mrow + 16 * (r >> 3);
    int v    = group * 4 + ((r >> 1) & 3);
    size_t fhi = ((size_t)mt * KC_TOT + kc) * 2;
    ap32[fhi * 256 + lane * 8 + v]       = ((unsigned)h1 << 16) | h0;
    ap32[(fhi + 1) * 256 + lane * 8 + v] = ((unsigned)l1 << 16) | l0;
}

// ---------------- pack B (weights) into WMMA bf16 hi/lo fragments + bias ----------------
// B frag (32x16 bf16): lane = n + 16*((K>>3)&1) ; K = (v>>2)*16 + (lane>>4)*8 + (v&3)*2 + parity.
__global__ void pack_B_kernel(const float* __restrict__ Wl, const float* __restrict__ Wr,
                              const float* __restrict__ b, int t, int phase,
                              unsigned int* __restrict__ bp32, float* __restrict__ bias,
                              int NT) {
    int tid = blockIdx.x * blockDim.x + threadIdx.x;
    if (tid >= 256 * NT * 16) return;
    int kp = tid & 255;
    int n  = tid >> 8;
    int k  = kp * 2;
    int kblk = k >> 7, kin = k & 127;
    int nin = n & 127, nblk = n >> 7;
    int gate;
    if (phase == 0) gate = (kblk < 2) ? (nblk ? 2 : 0) : (nblk ? 3 : 1);
    else            gate = (kblk < 2) ? 4 : 5;
    const float* M = (kblk & 1) ? Wr : Wl;
    size_t base = ((size_t)(gate * 2 + t) * DD + kin) * DD + nin;
    float w0 = M[base];
    float w1 = M[base + DD];
    unsigned short h0 = f2bf(w0), h1 = f2bf(w1);
    unsigned short l0 = f2bf(w0 - bf2f(h0)), l1 = f2bf(w1 - bf2f(h1));
    int kc = k >> 5;
    int kk = k & 31;
    int khalf = kk >> 4, rem = kk & 15;
    int v = khalf * 4 + ((rem >> 1) & 3);
    int lane = (n & 15) + 16 * (rem >> 3);
    int nt = n >> 4;
    size_t fhi = ((size_t)kc * NT + nt) * 2;
    bp32[fhi * 256 + lane * 8 + v]       = ((unsigned)h1 << 16) | h0;
    bp32[(fhi + 1) * 256 + lane * 8 + v] = ((unsigned)l1 << 16) | l0;
    if (kp == 0) {
        float bb;
        if (phase == 0) {
            int g0 = nblk ? 2 : 0, g1 = nblk ? 3 : 1;
            bb = b[(g0 * 2 + t) * DD + nin] + b[(g1 * 2 + t) * DD + nin];
        } else {
            bb = b[(8 + t) * DD + nin] + b[(10 + t) * DD + nin];
        }
        bias[n] = bb;
    }
}

// ---------------- WMMA GEMM with TDM/LDS double-buffered staging ----------------
// Block = 256 threads (8 waves), 128(M) x 128(N) tile. Wave = 32x64 (2x4 wmma tiles).
// Per K-chunk: A slice = 2-D tile (8 rows x 2KB, stride 32KB), B slice = contiguous 16KB.
// MODE 0: sigmoid -> z (cols<128) / hr = h*sigmoid(r) (cols>=128)
// MODE 1: tanh + GRU update -> out
template <int MODE>
__global__ __launch_bounds__(256) void gemm_wmma_kernel(
    const unsigned short* __restrict__ Apack,
    const unsigned short* __restrict__ Bpack,
    const float* __restrict__ bias,
    const float* __restrict__ hbuf,
    float* __restrict__ zbuf,
    float* __restrict__ hrbuf,
    float* __restrict__ outbuf,
    int NT, int nrows) {
    __shared__ unsigned short sh[2][16384];   // 2 x (16KB A + 16KB B)

    const int lane = threadIdx.x & 31;
    const int wave = threadIdx.x >> 5;
    const int wm = wave & 3, wn = wave >> 2;
    const int mb = blockIdx.x, nb = blockIdx.y;

    auto stage = [&](int buf, int kc) {
#if HAVE_TDM
        if (wave == 0) {
            unsigned base = (unsigned)(unsigned long long)(uintptr_t)&sh[buf][0];
            unsigned long long ga = (unsigned long long)(uintptr_t)Apack
                + ((size_t)mb * 8 * KC_TOT * 2 + (size_t)kc * 2) * 1024ull;
            tdm_load_tile(ga, base, 512u, 8u, 8192u);            // 8 x 2KB rows, 32KB stride
            unsigned long long gb = (unsigned long long)(uintptr_t)Bpack
                + (((size_t)kc * NT + nb * 8) * 2) * 1024ull;
            tdm_load_tile(gb, base + 16384u, 4096u, 1u, 4096u);  // contiguous 16KB
        }
#else
        for (int f = wave; f < 32; f += 8) {
            const unsigned short* s;
            unsigned short* d;
            if (f < 16) {
                int mt = f >> 1, hl = f & 1;
                s = Apack + (((size_t)(mb * 8 + mt) * KC_TOT + kc) * 2 + hl) * 512;
                d = &sh[buf][f * 512];
            } else {
                int bf = f - 16;
                s = Bpack + (((size_t)kc * NT + nb * 8) * 2 + bf) * 512;
                d = &sh[buf][8192 + bf * 512];
            }
            *(uint4*)(d + lane * 16)     = *(const uint4*)(s + lane * 16);
            *(uint4*)(d + lane * 16 + 8) = *(const uint4*)(s + lane * 16 + 8);
        }
#endif
    };

    v8f acc[2][4];
#pragma unroll
    for (int i = 0; i < 2; i++)
#pragma unroll
        for (int j = 0; j < 4; j++) acc[i][j] = (v8f){};

    stage(0, 0);

#pragma unroll 2
    for (int kc = 0; kc < KC_TOT; ++kc) {
        const int cbuf = kc & 1;
        if (kc + 1 < KC_TOT) stage((kc + 1) & 1, kc + 1);
#if HAVE_TDM
        if (wave == 0) {
            if (kc + 1 < KC_TOT) __builtin_amdgcn_s_wait_tensorcnt((short)2);
            else                 __builtin_amdgcn_s_wait_tensorcnt((short)0);
        }
#endif
        __syncthreads();      // buffer cbuf fully resident for all waves

        const unsigned short* As = &sh[cbuf][0];
        const unsigned short* Bs = &sh[cbuf][8192];
        v16bf ahi[2], alo[2], bhi[4], blo[4];
#pragma unroll
        for (int i = 0; i < 2; i++) {
            const unsigned short* pa = As + (wm * 2 + i) * 1024 + lane * 16;
            ahi[i] = *(const v16bf*)pa;
            alo[i] = *(const v16bf*)(pa + 512);
        }
#pragma unroll
        for (int j = 0; j < 4; j++) {
            const unsigned short* pb = Bs + (wn * 4 + j) * 1024 + lane * 16;
            bhi[j] = *(const v16bf*)pb;
            blo[j] = *(const v16bf*)(pb + 512);
        }
#pragma unroll
        for (int i = 0; i < 2; i++)
#pragma unroll
            for (int j = 0; j < 4; j++) {
                acc[i][j] = __builtin_amdgcn_wmma_f32_16x16x32_bf16(
                    false, ahi[i], false, bhi[j], (short)0, acc[i][j], false, false);
                acc[i][j] = __builtin_amdgcn_wmma_f32_16x16x32_bf16(
                    false, alo[i], false, bhi[j], (short)0, acc[i][j], false, false);
                acc[i][j] = __builtin_amdgcn_wmma_f32_16x16x32_bf16(
                    false, ahi[i], false, blo[j], (short)0, acc[i][j], false, false);
            }
        __syncthreads();      // all waves done reading cbuf before it is overwritten
    }

    // C layout: VGPR r -> M = r (lanes 0..15) / r+8 (lanes 16..31), N = lane&15
    int nlane = lane & 15;
    int mhalf = (lane >> 4) * 8;
#pragma unroll
    for (int i = 0; i < 2; i++) {
        int rowbase = mb * 128 + (wm * 2 + i) * 16 + mhalf;
#pragma unroll
        for (int r = 0; r < 8; r++) {
            int row = rowbase + r;
            if (row >= nrows) continue;
#pragma unroll
            for (int j = 0; j < 4; j++) {
                int col = (nb * 8 + wn * 4 + j) * 16 + nlane;
                float vv = acc[i][j][r] + bias[col];
                if (MODE == 0) {
                    float s = 1.0f / (1.0f + __expf(-vv));
                    if (col < DD) {
                        zbuf[(size_t)row * DD + col] = s;
                    } else {
                        int c = col - DD;
                        hrbuf[(size_t)row * DD + c] = hbuf[(size_t)row * DD + c] * s;
                    }
                } else {
                    float ht = tanhf(vv);
                    float z  = zbuf[(size_t)row * DD + col];
                    float h  = hbuf[(size_t)row * DD + col];
                    outbuf[(size_t)row * DD + col] = z * h + (1.0f - z) * ht;
                }
            }
        }
    }
}

extern "C" void kernel_launch(void* const* d_in, const int* in_sizes, int n_in,
                              void* d_out, int out_size, void* d_ws, size_t ws_size,
                              hipStream_t stream) {
    (void)in_sizes; (void)n_in; (void)out_size; (void)ws_size;
    const float* x_user = (const float*)d_in[0];
    const float* x_news = (const float*)d_in[1];
    const float* h_user = (const float*)d_in[2];
    const float* h_news = (const float*)d_in[3];
    const float* Wl     = (const float*)d_in[4];
    const float* Wr     = (const float*)d_in[5];
    const float* bb     = (const float*)d_in[6];
    const int* src_un   = (const int*)d_in[7];
    const int* dst_un   = (const int*)d_in[8];
    const int* src_nu   = (const int*)d_in[9];
    const int* dst_nu   = (const int*)d_in[10];
    float* out = (float*)d_out;          // [h_user_new | h_news_new]

    // ---- workspace carve ----
    char* p = (char*)d_ws;
    auto alloc = [&](size_t bytes) -> void* {
        void* r = (void*)p;
        p += (bytes + 255) & ~(size_t)255;
        return r;
    };
    const size_t fb = sizeof(float);
    float* deg_u   = (float*)alloc(NN * fb);
    float* deg_n   = (float*)alloc(NN * fb);
    float* agg_x_u = (float*)alloc((size_t)ND * fb);
    float* agg_x_n = (float*)alloc((size_t)ND * fb);
    float* agg_h_u = (float*)alloc((size_t)ND * fb);   // reused for agg_c_u
    float* agg_h_n = (float*)alloc((size_t)ND * fb);   // reused for agg_c_n
    float* z_u     = (float*)alloc((size_t)ND * fb);
    float* z_n     = (float*)alloc((size_t)ND * fb);
    float* hr_u    = (float*)alloc((size_t)ND * fb);
    float* hr_n    = (float*)alloc((size_t)ND * fb);
    const size_t APACK_BYTES = (size_t)MTILES * KC_TOT * 2 * 512 * 2;   // ~100 MB each
    unsigned short* Apack_u = (unsigned short*)alloc(APACK_BYTES);
    unsigned short* Apack_n = (unsigned short*)alloc(APACK_BYTES);
    unsigned short* Bzr_u = (unsigned short*)alloc((size_t)16 * 16 * 2 * 512 * 2);
    unsigned short* Bzr_n = (unsigned short*)alloc((size_t)16 * 16 * 2 * 512 * 2);
    unsigned short* Bc_u  = (unsigned short*)alloc((size_t)16 * 8 * 2 * 512 * 2);
    unsigned short* Bc_n  = (unsigned short*)alloc((size_t)16 * 8 * 2 * 512 * 2);
    float* bias_zr_u = (float*)alloc(256 * fb);
    float* bias_zr_n = (float*)alloc(256 * fb);
    float* bias_c_u  = (float*)alloc(128 * fb);
    float* bias_c_n  = (float*)alloc(128 * fb);

    // ---- zero accumulators ----
    (void)hipMemsetAsync(deg_u, 0, NN * fb, stream);
    (void)hipMemsetAsync(deg_n, 0, NN * fb, stream);
    (void)hipMemsetAsync(agg_x_u, 0, (size_t)ND * fb, stream);
    (void)hipMemsetAsync(agg_x_n, 0, (size_t)ND * fb, stream);
    (void)hipMemsetAsync(agg_h_u, 0, (size_t)ND * fb, stream);
    (void)hipMemsetAsync(agg_h_n, 0, (size_t)ND * fb, stream);

    const int TB = 256;
    const int degG = (EE + TB - 1) / TB;
    const long long scT = (long long)EE * 32;
    const int scG = (int)((scT + TB - 1) / TB);
    const int paG = (NROWS_PAD * 64 + TB - 1) / TB;

    // ---- degrees ----
    deg_kernel<<<degG, TB, 0, stream>>>(dst_un, deg_n, EE);
    deg_kernel<<<degG, TB, 0, stream>>>(dst_nu, deg_u, EE);

    // ---- aggregate x and h together (shared edge lists) ----
    scatter_add2_kernel<<<scG, TB, 0, stream>>>(x_user, h_user, src_un, dst_un,
                                                agg_x_n, agg_h_n, EE);
    scatter_add2_kernel<<<scG, TB, 0, stream>>>(x_news, h_news, src_nu, dst_nu,
                                                agg_x_u, agg_h_u, EE);

    // ---- pack weights (t: 0=news, 1=user per reference Wl[g, t]) ----
    pack_B_kernel<<<(256 * 256 + TB - 1) / TB, TB, 0, stream>>>(Wl, Wr, bb, 1, 0, (unsigned int*)Bzr_u, bias_zr_u, 16);
    pack_B_kernel<<<(256 * 256 + TB - 1) / TB, TB, 0, stream>>>(Wl, Wr, bb, 0, 0, (unsigned int*)Bzr_n, bias_zr_n, 16);
    pack_B_kernel<<<(256 * 128 + TB - 1) / TB, TB, 0, stream>>>(Wl, Wr, bb, 1, 1, (unsigned int*)Bc_u, bias_c_u, 8);
    pack_B_kernel<<<(256 * 128 + TB - 1) / TB, TB, 0, stream>>>(Wl, Wr, bb, 0, 1, (unsigned int*)Bc_n, bias_c_n, 8);

    // ---- pack A = [agg_x | x | agg_h | h] ----
    pack_A_kernel<<<paG, TB, 0, stream>>>(agg_x_u, deg_u,  (unsigned int*)Apack_u, 0,  NN);
    pack_A_kernel<<<paG, TB, 0, stream>>>(x_user,  nullptr,(unsigned int*)Apack_u, 4,  NN);
    pack_A_kernel<<<paG, TB, 0, stream>>>(agg_h_u, deg_u,  (unsigned int*)Apack_u, 8,  NN);
    pack_A_kernel<<<paG, TB, 0, stream>>>(h_user,  nullptr,(unsigned int*)Apack_u, 12, NN);
    pack_A_kernel<<<paG, TB, 0, stream>>>(agg_x_n, deg_n,  (unsigned int*)Apack_n, 0,  NN);
    pack_A_kernel<<<paG, TB, 0, stream>>>(x_news,  nullptr,(unsigned int*)Apack_n, 4,  NN);
    pack_A_kernel<<<paG, TB, 0, stream>>>(agg_h_n, deg_n,  (unsigned int*)Apack_n, 8,  NN);
    pack_A_kernel<<<paG, TB, 0, stream>>>(h_news,  nullptr,(unsigned int*)Apack_n, 12, NN);

    // ---- GEMM 1: produce z and hr = h * sigmoid(r) ----
    gemm_wmma_kernel<0><<<dim3(391, 2), TB, 0, stream>>>(Apack_u, Bzr_u, bias_zr_u,
        h_user, z_u, hr_u, nullptr, 16, NN);
    gemm_wmma_kernel<0><<<dim3(391, 2), TB, 0, stream>>>(Apack_n, Bzr_n, bias_zr_n,
        h_news, z_n, hr_n, nullptr, 16, NN);

    // ---- aggregate hr (reuse agg_h buffers) ----
    (void)hipMemsetAsync(agg_h_u, 0, (size_t)ND * fb, stream);
    (void)hipMemsetAsync(agg_h_n, 0, (size_t)ND * fb, stream);
    scatter_add_kernel<<<scG, TB, 0, stream>>>(hr_u, src_un, dst_un, agg_h_n, EE);
    scatter_add_kernel<<<scG, TB, 0, stream>>>(hr_n, src_nu, dst_nu, agg_h_u, EE);

    // ---- repack K-chunks 8..15 = [agg_c | hr] ----
    pack_A_kernel<<<paG, TB, 0, stream>>>(agg_h_u, deg_u,  (unsigned int*)Apack_u, 8,  NN);
    pack_A_kernel<<<paG, TB, 0, stream>>>(hr_u,    nullptr,(unsigned int*)Apack_u, 12, NN);
    pack_A_kernel<<<paG, TB, 0, stream>>>(agg_h_n, deg_n,  (unsigned int*)Apack_n, 8,  NN);
    pack_A_kernel<<<paG, TB, 0, stream>>>(hr_n,    nullptr,(unsigned int*)Apack_n, 12, NN);

    // ---- GEMM 2: tanh + GRU update -> d_out ----
    gemm_wmma_kernel<1><<<dim3(391, 1), TB, 0, stream>>>(Apack_u, Bc_u, bias_c_u,
        h_user, z_u, nullptr, out, 8, NN);
    gemm_wmma_kernel<1><<<dim3(391, 1), TB, 0, stream>>>(Apack_n, Bc_n, bias_c_n,
        h_news, z_n, nullptr, out + (size_t)ND, 8, NN);
}
